// MultiHeadSelfAttention_44942537785799
// MI455X (gfx1250) — matrix operand order repaired and checked by reading, not compiled, not run on hardware
//
#include <hip/hip_runtime.h>
#include <hip/hip_bf16.h>
#include <math.h>

// ---------------------------------------------------------------------------
// MI455X (gfx1250) fused multi-head self-attention.
// Roofline: ~70 GFLOP vs ~100 MB HBM traffic @ 23.3 TB/s -> bandwidth bound.
// bf16 WMMA (v_wmma_f32_16x16x32_bf16, f32 accum); flash attention keeps the
// 2048x2048 score matrix out of HBM. wave32 A/B fragments use contiguous k = 16*(lane/16) + e.
// ---------------------------------------------------------------------------

typedef __attribute__((ext_vector_type(16))) __bf16 v16bf;
typedef __attribute__((ext_vector_type(8)))  float  v8f;
typedef __attribute__((ext_vector_type(4)))  float  vf4;

#define B_BATCH 2
#define T_SEQ   2048
#define NH      16
#define HD      64
#define DIM     1024
#define DIM3    3072
#define MROWS   4096            // B_BATCH * T_SEQ
#define SCALE_Q 0.125f          // 64^-0.5

// A-fragment k order used in this file (B matches it): lane L holds row M=L%16;
// element e maps to K = e + (L/16)*16  (one contiguous run of 16).
__device__ __forceinline__ int a_kmap(int e, int half) {
    return e + half * 16;
}

// Load an A fragment from 32 consecutive K values at p (row base + k0),
// using one aligned 16-float run per lane half -> four global_load_b128.
__device__ __forceinline__ v16bf load_a_frag(const float* p, int half) {
    const float* pa = p + half * 16;
    vf4 q0 = *(const vf4*)(pa);
    vf4 q1 = *(const vf4*)(pa + 4);
    vf4 q2 = *(const vf4*)(pa + 8);
    vf4 q3 = *(const vf4*)(pa + 12);
    v16bf a;
#pragma unroll
    for (int i = 0; i < 4; ++i) {
        a[i]      = (__bf16)q0[i];
        a[4 + i]  = (__bf16)q1[i];
        a[8 + i]  = (__bf16)q2[i];
        a[12 + i] = (__bf16)q3[i];
    }
    return a;
}

// ---------------------------------------------------------------------------
// Wave-level GEMM: C[M,N] = A[M,K] * B[K,N] (+bias). 16x64 tile per wave:
// one A fragment feeds 4 WMMAs per 32-wide K step (4x compute per A byte).
// ---------------------------------------------------------------------------
template <bool HAS_BIAS>
__global__ void gemm_bf16_kernel(const float* __restrict__ A,
                                 const float* __restrict__ B,
                                 const float* __restrict__ bias,
                                 float* __restrict__ C,
                                 int M, int N, int K) {
    const int lane = threadIdx.x & 31;
    const int wave = threadIdx.x >> 5;
    const int half = lane >> 4;
    const int n16  = lane & 15;

    const int m0 = blockIdx.x * 16;
    const int n0 = (blockIdx.y * 8 + wave) * 64;
    if (m0 >= M || n0 >= N) return;

    const float* arow = A + (size_t)(m0 + n16) * K;
    v8f acc[4] = {};

    for (int k0 = 0; k0 < K; k0 += 32) {
        const v16bf a = load_a_frag(arow + k0, half);
        // Prefetch next K-slice of the streamed weight panel into cache.
        if (k0 + 32 < K)
            __builtin_prefetch(&B[(size_t)(k0 + 32 + half * 16) * N + n0 + n16], 0, 3);
#pragma unroll
        for (int nb = 0; nb < 4; ++nb) {
            v16bf b;
#pragma unroll
            for (int e = 0; e < 16; ++e) {
                // B fragment (KxN): lane holds col, K = k0 + e + half*16
                b[e] = (__bf16)B[(size_t)(k0 + e + half * 16) * N + n0 + nb * 16 + n16];
            }
            acc[nb] = __builtin_amdgcn_wmma_f32_16x16x32_bf16(
                false, a, false, b, (short)0, acc[nb], false, false);
        }
    }

#pragma unroll
    for (int nb = 0; nb < 4; ++nb) {
        const float bv = HAS_BIAS ? bias[n0 + nb * 16 + n16] : 0.0f;
#pragma unroll
        for (int r = 0; r < 8; ++r) {
            const int row = m0 + r + half * 8;        // C/D layout
            C[(size_t)row * N + n0 + nb * 16 + n16] = acc[nb][r] + bv;
        }
    }
}

// ---------------------------------------------------------------------------
// RoPE on Q and K slices of the QKV buffer, in place; folds softmax scale
// into Q. One thread per (b, part, h, t, d<32) rotation pair.
// ---------------------------------------------------------------------------
__global__ void rope_kernel(float* __restrict__ qkv) {
    const int idx  = blockIdx.x * blockDim.x + threadIdx.x;
    const int d    = idx & 31;
    const int t    = (idx >> 5) & (T_SEQ - 1);
    const int h    = (idx >> 16) & (NH - 1);
    const int part = (idx >> 20) & 1;                 // 0 = q, 1 = k
    const int b    = idx >> 21;

    const float freq = __expf(-9.210340371976184f * (float)d * (1.0f / 32.0f));
    const float ang  = (float)t * freq;
    float sn, cs;
    __sincosf(ang, &sn, &cs);

    const size_t base = ((size_t)(b * T_SEQ + t)) * DIM3 + part * DIM + h * HD;
    const float x1 = qkv[base + d];
    const float x2 = qkv[base + 32 + d];
    const float sc = (part == 0) ? SCALE_Q : 1.0f;
    qkv[base + d]      = (x1 * cs - x2 * sn) * sc;
    qkv[base + 32 + d] = (x1 * sn + x2 * cs) * sc;
}

// ---------------------------------------------------------------------------
// Flash attention: one 16-row q-tile per wave, key blocks of 32, online
// softmax. S and P*V via bf16 WMMA. P bounces through per-wave LDS to go
// from C-layout to A-layout (wave-local DS ordering + s_wait_dscnt; no
// cross-wave barrier, so causal loop-count divergence between waves is safe).
// ---------------------------------------------------------------------------
__global__ void attn_kernel(const float* __restrict__ qkv,
                            float* __restrict__ attn_out) {
    const int lane = threadIdx.x & 31;
    const int wave = threadIdx.x >> 5;
    const int half = lane >> 4;
    const int n16  = lane & 15;

    const int h  = blockIdx.y;
    const int b  = blockIdx.z;
    const int q0 = (blockIdx.x * 8 + wave) * 16;
    const size_t rowbase = (size_t)b * T_SEQ;

    const float* Qp = qkv + rowbase * DIM3 + 0 * DIM + h * HD;
    const float* Kp = qkv + rowbase * DIM3 + 1 * DIM + h * HD;
    const float* Vp = qkv + rowbase * DIM3 + 2 * DIM + h * HD;

    __shared__ __bf16 lds_p[8][16 * 32];              // 1 KB per wave
    __bf16* myp = &lds_p[wave][0];

    // Q fragments (A layout), head-dim chunks 0..31 and 32..63.
    const float* qrowp = Qp + (size_t)(q0 + n16) * DIM3;
    const v16bf aq0 = load_a_frag(qrowp, half);
    const v16bf aq1 = load_a_frag(qrowp + 32, half);

    float mrow[8], lrow[8];
    v8f acc[4] = {};
#pragma unroll
    for (int r = 0; r < 8; ++r) { mrow[r] = -3.0e38f; lrow[r] = 0.0f; }

    const int q_last = q0 + 15;
    for (int j0 = 0; j0 <= q_last; j0 += 32) {
        // ---- S = Q * K^T over a 16x32 block of scores --------------------
        v8f s[2];
#pragma unroll
        for (int sub = 0; sub < 2; ++sub) {
            // B fragment for K^T: lane = key col, K index = head dim.
            // 16 consecutive, 16B-aligned floats per chunk -> b128 loads.
            const float* krow = Kp + (size_t)(j0 + sub * 16 + n16) * DIM3 + half * 16;
            v16bf bk0, bk1;
#pragma unroll
            for (int i = 0; i < 4; ++i) {
                vf4 c0 = *(const vf4*)(krow + 4 * i);
                vf4 c1 = *(const vf4*)(krow + 32 + 4 * i);
#pragma unroll
                for (int j = 0; j < 4; ++j) {
                    bk0[4 * i + j] = (__bf16)c0[j];
                    bk1[4 * i + j] = (__bf16)c1[j];
                }
            }
            v8f z = {};
            z = __builtin_amdgcn_wmma_f32_16x16x32_bf16(
                false, aq0, false, bk0, (short)0, z, false, false);
            z = __builtin_amdgcn_wmma_f32_16x16x32_bf16(
                false, aq1, false, bk1, (short)0, z, false, false);
            s[sub] = z;
        }

        // ---- causal mask + online softmax (rows live in 16-lane halves) --
#pragma unroll
        for (int r = 0; r < 8; ++r) {
            const int qrow = q0 + r + half * 8;
            float s0 = s[0][r]; if (j0 + n16 > qrow)      s0 = -1.0e30f;
            float s1 = s[1][r]; if (j0 + 16 + n16 > qrow) s1 = -1.0e30f;

            float bm = fmaxf(s0, s1);
#pragma unroll
            for (int off = 8; off; off >>= 1)
                bm = fmaxf(bm, __shfl_xor(bm, off, 32));

            const float nm    = fmaxf(mrow[r], bm);
            const float alpha = __expf(mrow[r] - nm);
            const float p0    = __expf(s0 - nm);
            const float p1    = __expf(s1 - nm);

            float rs = p0 + p1;
#pragma unroll
            for (int off = 8; off; off >>= 1)
                rs += __shfl_xor(rs, off, 32);

            lrow[r] = lrow[r] * alpha + rs;
            mrow[r] = nm;
#pragma unroll
            for (int nb = 0; nb < 4; ++nb) acc[nb][r] *= alpha;

            // P tile to LDS in plain (row, col) order (C layout source).
            myp[(r + half * 8) * 32 + n16]      = (__bf16)p0;
            myp[(r + half * 8) * 32 + 16 + n16] = (__bf16)p1;
        }

        // Wave-local LDS RAW: DS ops are in-order per wave; wait for data.
        asm volatile("s_wait_dscnt 0x0" ::: "memory");

        // ---- reload P in A layout ---------------------------------------
        v16bf pa;
#pragma unroll
        for (int e = 0; e < 16; ++e)
            pa[e] = myp[n16 * 32 + a_kmap(e, half)];

        // ---- O += P * V --------------------------------------------------
#pragma unroll
        for (int nb = 0; nb < 4; ++nb) {
            v16bf bv;
#pragma unroll
            for (int e = 0; e < 16; ++e) {
                const int j = e + half * 16;           // key within block
                bv[e] = (__bf16)Vp[(size_t)(j0 + j) * DIM3 + nb * 16 + n16];
            }
            acc[nb] = __builtin_amdgcn_wmma_f32_16x16x32_bf16(
                false, pa, false, bv, (short)0, acc[nb], false, false);
        }
    }

    // ---- normalize, store (b, t, h*64+d) so proj GEMM reads row-major ----
#pragma unroll
    for (int r = 0; r < 8; ++r) {
        const float inv  = 1.0f / lrow[r];
        const int   qrow = q0 + r + half * 8;
#pragma unroll
        for (int nb = 0; nb < 4; ++nb) {
            attn_out[(rowbase + qrow) * (size_t)DIM + h * HD + nb * 16 + n16] =
                acc[nb][r] * inv;
        }
    }
}

// ---------------------------------------------------------------------------
// Launch: qkv GEMM -> RoPE -> flash attention -> output projection (+bias).
// Workspace: qkv f32 (4096x3072) + attn f32 (4096x1024) = 64 MiB.
// ---------------------------------------------------------------------------
extern "C" void kernel_launch(void* const* d_in, const int* in_sizes, int n_in,
                              void* d_out, int out_size, void* d_ws, size_t ws_size,
                              hipStream_t stream) {
    const float* x      = (const float*)d_in[0];
    const float* w_qkv  = (const float*)d_in[1];
    const float* w_proj = (const float*)d_in[2];
    const float* b_proj = (const float*)d_in[3];
    float* out  = (float*)d_out;

    float* qkv  = (float*)d_ws;                        // 4096*3072 f32
    float* attn = qkv + (size_t)MROWS * DIM3;          // 4096*1024 f32

    dim3 blk(256);

    // QKV = x @ w_qkv   (M=4096, N=3072, K=1024); 16x64 tile/wave.
    gemm_bf16_kernel<false><<<dim3(MROWS / 16, DIM3 / (64 * 8)), blk, 0, stream>>>(
        x, w_qkv, nullptr, qkv, MROWS, DIM3, DIM);

    // RoPE on q,k (+ fold softmax scale into q)
    const int rope_threads = B_BATCH * 2 * NH * T_SEQ * 32;   // 4,194,304
    rope_kernel<<<rope_threads / 256, blk, 0, stream>>>(qkv);

    // Flash attention: 8 q-tiles (waves) per block, grid (T/128, NH, B)
    attn_kernel<<<dim3(T_SEQ / 128, NH, B_BATCH), blk, 0, stream>>>(qkv, attn);

    // out = attn @ w_proj + b_proj  (M=4096, N=1024, K=1024)
    gemm_bf16_kernel<true><<<dim3(MROWS / 16, DIM / (64 * 8)), blk, 0, stream>>>(
        attn, w_proj, b_proj, out, MROWS, DIM, DIM);
}
